// MultiHeadSelfAttention_8254927143408
// MI455X (gfx1250) — compile-verified, matrix operand order checked by reading, not run on hardware
//
#include <hip/hip_runtime.h>

typedef __attribute__((ext_vector_type(16))) __bf16 v16bf;
typedef __attribute__((ext_vector_type(8)))  __bf16 v8bf;
typedef __attribute__((ext_vector_type(4)))  __bf16 v4bf;
typedef __attribute__((ext_vector_type(8)))  float  v8f;

#define WMMA_BF16(a, b, c) \
    __builtin_amdgcn_wmma_f32_16x16x32_bf16(false, (a), false, (b), (short)0, (c), false, false)

static constexpr int Bc  = 2;
static constexpr int Sc  = 2048;
static constexpr int Dc  = 1024;
static constexpr int Hc  = 16;
static constexpr int HDc = 64;
static constexpr float SCALEc = 0.125f;   // 64^-0.5

// fp32 -> bf16: native hardware convert (RNE on CDNA5)
__device__ __forceinline__ __bf16 f2bf(float f) { return (__bf16)f; }

__device__ __forceinline__ v4bf f4tobf(float4 f) {
    v4bf h;
    h[0] = f2bf(f.x); h[1] = f2bf(f.y); h[2] = f2bf(f.z); h[3] = f2bf(f.w);
    return h;
}

// One 16-lane fragment (ISA 7.12.2, 16-bit 16x32 layout): lane hi-half selects
// K {hi*8..+7} and {16+hi*8..+7} -> two contiguous 16B runs -> 2x ds_load_b128.
__device__ __forceinline__ v16bf load_frag(const __bf16* rowPtr, int hi) {
    const __bf16* p = rowPtr + hi * 8;
    v8bf lo = *(const v8bf*)(p);
    v8bf hv = *(const v8bf*)(p + 16);
    return __builtin_shufflevector(lo, hv, 0, 1, 2, 3, 4, 5, 6, 7,
                                   8, 9, 10, 11, 12, 13, 14, 15);
}

// ---------------------------------------------------------------------------
// GEMM: C[M,N] = A[M,K] @ B[K,N] + bias[N]. Block tile 128x64, K-step 64,
// double-buffered LDS. 8 waves; wave w owns rows [w*16,w*16+16) x 64 cols
// (4 accumulators, A-fragment reused 4x). B transposed in LDS ([n][k]).
// MODE 0: row-major store. MODE 1: scatter q/k/v into [B,H,S,HD] buffers.
// ---------------------------------------------------------------------------
constexpr int GLD = 72;                             // 64 + 8 pad (144 B rows)

__device__ __forceinline__ void gemm_load_tile(
    __bf16 (*As)[GLD], __bf16 (*Bs)[GLD],
    const float* __restrict__ A, const float* __restrict__ Bm,
    int N, int K, int tileM, int tileN, int k0, int tid)
{
    #pragma unroll
    for (int i = 0; i < 8; ++i) {                   // 128x64 A tile (vec4)
        int idx = tid + i * 256;                    // 2048 float4s
        int r = idx >> 4, c4 = idx & 15;
        float4 f = *(const float4*)&A[(size_t)(tileM + r) * K + k0 + c4 * 4];
        *(v4bf*)&As[r][c4 * 4] = f4tobf(f);
    }
    #pragma unroll
    for (int i = 0; i < 4; ++i) {                   // 64x64 B tile (vec4), transpose
        int idx = tid + i * 256;                    // 1024 float4s
        int r = idx >> 4, c4 = idx & 15;            // r = k row, c4*4 = n col
        float4 f = *(const float4*)&Bm[(size_t)(k0 + r) * N + tileN + c4 * 4];
        Bs[c4 * 4 + 0][r] = f2bf(f.x);
        Bs[c4 * 4 + 1][r] = f2bf(f.y);
        Bs[c4 * 4 + 2][r] = f2bf(f.z);
        Bs[c4 * 4 + 3][r] = f2bf(f.w);
    }
}

template <int MODE>
__global__ __launch_bounds__(256) void gemm_bf16_wmma(
    const float* __restrict__ A, const float* __restrict__ Bm,
    const float* __restrict__ bias, float* __restrict__ C,
    float* __restrict__ Cq, float* __restrict__ Ck, float* __restrict__ Cv,
    int M, int N, int K)
{
    constexpr int BK = 64;
    __shared__ __align__(16) __bf16 As[2][128][GLD];   // [m][k]
    __shared__ __align__(16) __bf16 Bs[2][64][GLD];    // [n][k] (transposed)

    const int tid  = threadIdx.x;
    const int wave = tid >> 5;
    const int lane = tid & 31;
    const int l    = lane & 15;
    const int hi   = lane >> 4;
    const int wRow = wave * 16;
    const int tileM = blockIdx.y * 128;
    const int tileN = blockIdx.x * 64;

    v8f acc[4] = {};

    gemm_load_tile(As[0], Bs[0], A, Bm, N, K, tileM, tileN, 0, tid);

    int buf = 0;
    for (int k0 = 0; k0 < K; k0 += BK) {
        __syncthreads();
        if (k0 + 2 * BK < K) {                       // prefetch block after next
            int pr = tid >> 1;
            __builtin_prefetch(&A[(size_t)(tileM + pr) * K + k0 + 2 * BK + (tid & 1) * 32], 0, 0);
        }
        if (k0 + BK < K)
            gemm_load_tile(As[buf ^ 1], Bs[buf ^ 1], A, Bm, N, K, tileM, tileN, k0 + BK, tid);

        #pragma unroll
        for (int kb = 0; kb < BK; kb += 32) {
            v16bf af = load_frag(&As[buf][wRow + l][kb], hi);
            #pragma unroll
            for (int t = 0; t < 4; ++t) {
                v16bf bf = load_frag(&Bs[buf][t * 16 + l][kb], hi);
                acc[t] = WMMA_BF16(af, bf, acc[t]);
            }
        }
        buf ^= 1;
    }

    if (MODE == 0) {
        #pragma unroll
        for (int t = 0; t < 4; ++t) {
            #pragma unroll
            for (int r = 0; r < 8; ++r) {
                int mG = tileM + wRow + r + 8 * hi;
                int nG = tileN + t * 16 + l;
                C[(size_t)mG * N + nG] = acc[t][r] + bias[nG];
            }
        }
    } else {
        // Block-uniform scatter: tileN % 64 == 0 -> sel/head fixed per block;
        // tileM % 128 == 0 and S = 2048 -> batch fixed per block.
        const int sel = tileN >> 10;
        const int hh  = (tileN & 1023) >> 6;
        const int bb  = tileM >> 11;
        const int ssB = tileM & 2047;
        float* dst = (sel == 0 ? Cq : sel == 1 ? Ck : Cv)
                   + (((size_t)(bb * Hc + hh)) * Sc + ssB) * HDc;
        #pragma unroll
        for (int t = 0; t < 4; ++t) {
            #pragma unroll
            for (int r = 0; r < 8; ++r) {
                int mloc = wRow + r + 8 * hi;
                int hd   = t * 16 + l;
                dst[(size_t)mloc * HDc + hd] = acc[t][r] + bias[tileN + hd];
            }
        }
    }
}

// ---------------------------------------------------------------------------
// Flash attention: one block = one (b, h, 64-row q-tile). Online softmax
// (4 threads per row), analytic causal mask. V transposed in LDS so P@V
// B-fragments are contiguous. O accumulators in VGPRs.
// ---------------------------------------------------------------------------
__global__ __launch_bounds__(256) void attn_wmma(
    const float* __restrict__ Q, const float* __restrict__ Kt,
    const float* __restrict__ Vt, float* __restrict__ Ao)
{
    constexpr int LD = 72;
    __shared__ __align__(16) __bf16 Qs[64][LD];     // [q][d], pre-scaled
    __shared__ __align__(16) __bf16 Ks[64][LD];     // [key][d]
    __shared__ __align__(16) __bf16 VsT[64][LD];    // [d][key] (transposed)
    __shared__ __align__(16) __bf16 Ps[64][LD];     // [q][key]
    __shared__ float  Sb[64][66];
    __shared__ float  pmax[64][4];
    __shared__ float  psum[64][4];
    __shared__ float  row_m[64];
    __shared__ float  row_l[64];
    __shared__ float  row_a[64];

    const int tid  = threadIdx.x;
    const int wave = tid >> 5;
    const int lane = tid & 31;
    const int l    = lane & 15;
    const int hi   = lane >> 4;
    const int wRow = (wave >> 1) * 16;
    const int wCol = (wave & 1) * 32;
    const int srow = tid >> 2;                      // softmax: row (0..63)
    const int squad = tid & 3;                      //          16-col quarter

    const int bh    = blockIdx.y;                   // b*H + h
    const int b     = bh / Hc;
    const int h     = bh % Hc;
    const int qTile = blockIdx.x;
    const int qBase = qTile * 64;
    const size_t base = ((size_t)bh) * Sc * HDc;

    // 64x64 Q tile: fully contiguous 4096 floats
    {
        const float* Qp = Q + base + (size_t)qBase * HDc;
        #pragma unroll
        for (int i = 0; i < 4; ++i) {
            int idx = tid + i * 256;                // float4 index
            int r = idx >> 4, c4 = idx & 15;
            float4 f = *(const float4*)&Qp[(size_t)idx * 4];
            f.x *= SCALEc; f.y *= SCALEc; f.z *= SCALEc; f.w *= SCALEc;
            *(v4bf*)&Qs[r][c4 * 4] = f4tobf(f);
        }
    }
    if (tid < 64) { row_m[tid] = -1e30f; row_l[tid] = 0.0f; }
    __syncthreads();

    v8f o0 = {};
    v8f o1 = {};

    for (int j = 0; j <= qTile; ++j) {
        const int kBase = j * 64;
        const float* Kp = Kt + base + (size_t)kBase * HDc;
        const float* Vp = Vt + base + (size_t)kBase * HDc;
        #pragma unroll
        for (int i = 0; i < 4; ++i) {
            int idx = tid + i * 256;
            int r = idx >> 4, c4 = idx & 15;
            float4 fk = *(const float4*)&Kp[(size_t)idx * 4];
            *(v4bf*)&Ks[r][c4 * 4] = f4tobf(fk);
            float4 fv = *(const float4*)&Vp[(size_t)idx * 4];
            VsT[c4 * 4 + 0][r] = f2bf(fv.x);
            VsT[c4 * 4 + 1][r] = f2bf(fv.y);
            VsT[c4 * 4 + 2][r] = f2bf(fv.z);
            VsT[c4 * 4 + 3][r] = f2bf(fv.w);
        }
        __syncthreads();

        // ---- scores: S = Q @ K^T (contraction over d) ----
        v8f s0 = {};
        v8f s1 = {};
        #pragma unroll
        for (int kb = 0; kb < 64; kb += 32) {
            v16bf af = load_frag(&Qs[wRow + l][kb], hi);
            v16bf b0 = load_frag(&Ks[wCol + l][kb], hi);       // B[k=d][n=key]
            v16bf b1 = load_frag(&Ks[wCol + 16 + l][kb], hi);
            s0 = WMMA_BF16(af, b0, s0);
            s1 = WMMA_BF16(af, b1, s1);
        }

        // ---- causal mask + spill ----
        #pragma unroll
        for (int r = 0; r < 8; ++r) {
            int mloc = wRow + r + 8 * hi;
            int qg   = qBase + mloc;
            int n0   = wCol + l;
            int n1   = n0 + 16;
            Sb[mloc][n0] = (kBase + n0 > qg) ? -1e30f : s0[r];
            Sb[mloc][n1] = (kBase + n1 > qg) ? -1e30f : s1[r];
        }
        __syncthreads();

        // ---- online softmax: 4 threads per row, 16 cols each ----
        {
            const int c0 = squad * 16;
            float mx = -1e30f;
            #pragma unroll
            for (int c = 0; c < 16; ++c) mx = fmaxf(mx, Sb[srow][c0 + c]);
            pmax[srow][squad] = mx;
        }
        __syncthreads();
        {
            const int c0 = squad * 16;
            float mx = fmaxf(fmaxf(pmax[srow][0], pmax[srow][1]),
                             fmaxf(pmax[srow][2], pmax[srow][3]));
            mx = fmaxf(mx, row_m[srow]);
            float ls = 0.0f;
            #pragma unroll
            for (int c = 0; c < 16; c += 4) {
                v4bf pv;
                #pragma unroll
                for (int u = 0; u < 4; ++u) {
                    float p = __expf(Sb[srow][c0 + c + u] - mx);
                    pv[u] = f2bf(p);
                    ls += p;
                }
                *(v4bf*)&Ps[srow][c0 + c] = pv;
            }
            psum[srow][squad] = ls;
            if (squad == 0) {
                float al = __expf(row_m[srow] - mx);
                row_m[srow] = mx;
                row_a[srow] = al;
            }
        }
        __syncthreads();
        if (squad == 0) {
            row_l[srow] = row_l[srow] * row_a[srow] +
                          (psum[srow][0] + psum[srow][1] +
                           psum[srow][2] + psum[srow][3]);
        }

        // ---- rescale O, accumulate P @ V ----
        #pragma unroll
        for (int r = 0; r < 8; ++r) {
            float a_r = row_a[wRow + r + 8 * hi];
            o0[r] *= a_r;
            o1[r] *= a_r;
        }
        #pragma unroll
        for (int kb = 0; kb < 64; kb += 32) {       // contraction over key
            v16bf af = load_frag(&Ps[wRow + l][kb], hi);
            v16bf b0 = load_frag(&VsT[wCol + l][kb], hi);      // B[k=key][n=d]
            v16bf b1 = load_frag(&VsT[wCol + 16 + l][kb], hi);
            o0 = WMMA_BF16(af, b0, o0);
            o1 = WMMA_BF16(af, b1, o1);
        }
        __syncthreads();
    }

    // ---- normalize, write out in [B, S, D] layout ----
    #pragma unroll
    for (int r = 0; r < 8; ++r) {
        int mloc = wRow + r + 8 * hi;
        int s    = qBase + mloc;
        float invl = 1.0f / row_l[mloc];
        size_t rowOff = ((size_t)b * Sc + s) * Dc + (size_t)h * HDc;
        Ao[rowOff + wCol + l]      = o0[r] * invl;
        Ao[rowOff + wCol + 16 + l] = o1[r] * invl;
    }
}

// ---------------------------------------------------------------------------
// Host-side launch
// ---------------------------------------------------------------------------
extern "C" void kernel_launch(void* const* d_in, const int* in_sizes, int n_in,
                              void* d_out, int out_size, void* d_ws, size_t ws_size,
                              hipStream_t stream) {
    (void)in_sizes; (void)n_in; (void)out_size; (void)ws_size;

    const float* x     = (const float*)d_in[0];
    // d_in[1] = causal mask; computed analytically in-kernel.
    const float* w_qkv = (const float*)d_in[2];
    const float* b_qkv = (const float*)d_in[3];
    const float* w_out = (const float*)d_in[4];
    const float* b_out = (const float*)d_in[5];

    float* out  = (float*)d_out;                              // [B,S,D]
    float* kout = out  + (size_t)Bc * Sc * Dc;                // [B,H,S,HD]
    float* vout = kout + (size_t)Bc * Hc * Sc * HDc;          // [B,H,S,HD]

    float* q_ws    = (float*)d_ws;                            // [B,H,S,HD]
    float* attn_ws = q_ws + (size_t)Bc * Hc * Sc * HDc;       // [B,S,D]

    const int M = Bc * Sc;                                    // 4096

    // 1) QKV projection: [4096,1024] @ [1024,3072] + bias -> scatter q/k/v
    gemm_bf16_wmma<1><<<dim3((3 * Dc) / 64, M / 128), 256, 0, stream>>>(
        x, w_qkv, b_qkv, nullptr, q_ws, kout, vout, M, 3 * Dc, Dc);

    // 2) causal flash attention per (b,h,q-tile)
    attn_wmma<<<dim3(Sc / 64, Bc * Hc), 256, 0, stream>>>(q_ws, kout, vout, attn_ws);

    // 3) output projection: [4096,1024] @ [1024,1024] + bias -> out
    gemm_bf16_wmma<0><<<dim3(Dc / 64, M / 128), 256, 0, stream>>>(
        attn_ws, w_out, b_out, out, nullptr, nullptr, nullptr, M, Dc, Dc);
}
